// EdgeProcessor_78915729097028
// MI455X (gfx1250) — matrix operand order repaired and checked by reading, not compiled, not run on hardware
//
#include <hip/hip_runtime.h>
#include <hip/hip_bf16.h>

typedef __attribute__((ext_vector_type(16))) _Float16 v16h;
typedef __attribute__((ext_vector_type(8)))  float    v8f;

#define N_EDGES   65536
#define N_TILES   (N_EDGES / 16)
#define INV_SQRT3 0.5773502691896258f
#define BN_EPS    1e-5f

// Packed-weight fragment regions (units: halves). One B fragment = 32 lanes x 16 halves.
#define FRAG   512
#define SS_CH  65          // K = 65*32   (kidx = i*32+j)
#define VV_CH  48          // K = 3*512   (kidx = x*512 + i*16 + j), weights pre-scaled by 1/sqrt(3)
#define SV_CH  33          // K = 1056    (kidx = i*16+j, i padded 65->66)
#define VS_CH  32          // K = 32*32   (kidx = i*32+j)
#define SS_OFF 0
#define SS_SZ  (SS_CH * 3 * FRAG)
#define VV_OFF (SS_OFF + SS_SZ)
#define VV_SZ  (VV_CH * 3 * FRAG)
#define SV_OFF (VV_OFF + VV_SZ)
#define SV_SZ  (SV_CH * FRAG)
#define VS_OFF (SV_OFF + SV_SZ)
#define VS_SZ  (VS_CH * FRAG)
#define PK_TOTAL (VS_OFF + VS_SZ)          // 206848 halves

// Workspace byte offsets
#define OFF_SCAL  425984ull                               // after packed weights (413696B, padded)
#define OFF_VOUT  (OFF_SCAL + (size_t)N_EDGES * 32 * 4)   // scalars buf E x 32 f32
#define OFF_PART  (OFF_VOUT + (size_t)N_EDGES * 48 * 4)   // vout buf E x [x][k] f32
#define OFF_STATS (OFF_PART + (size_t)N_TILES * 80 * 4)   // per-block partials

#define WMMA_F16(A, B, C) \
  __builtin_amdgcn_wmma_f32_16x16x32_f16(false, (A), false, (B), (short)0, (C), false, false)

// ---------------------------------------------------------------------------
// Kernel 0: convert f32 weights -> f16, permuted into the wave32 WMMA B-layout
// (half h<8 -> K=h, h>=8 -> K=h+8, +8 if lane>=16; N = lane%16)
// ---------------------------------------------------------------------------
__global__ void k_pack(const float* __restrict__ Wss, const float* __restrict__ Wvv,
                       const float* __restrict__ Wsv, const float* __restrict__ Wvs,
                       _Float16* __restrict__ pk)
{
  int idx = blockIdx.x * blockDim.x + threadIdx.x;
  if (idx >= PK_TOTAL) return;

  int region, local;
  if (idx < VV_OFF)      { region = 0; local = idx - SS_OFF; }
  else if (idx < SV_OFF) { region = 1; local = idx - VV_OFF; }
  else if (idx < VS_OFF) { region = 2; local = idx - SV_OFF; }
  else                   { region = 3; local = idx - VS_OFF; }

  int h    = local & 15;
  int lane = (local >> 4) & 31;
  int cn   = local >> 9;
  int Kl   = h + ((h < 8) ? 0 : 8) + ((lane >= 16) ? 8 : 0);   // B-fragment K index within the 32-chunk
  int kn   = lane & 15;                     // N index

  float val = 0.0f;
  if (region == 0) {                        // W_ss (65,32,48): chunk c = i, Kl = j
    int c = cn / 3, nt = cn - 3 * c;
    val = Wss[(c * 32 + Kl) * 48 + nt * 16 + kn];
  } else if (region == 1) {                 // W_vv0 (32,16,48), x folded into K, scaled
    int c = cn / 3, nt = cn - 3 * c;
    int c16 = c & 15;
    int i = 2 * c16 + (Kl >> 4);
    int j = Kl & 15;
    val = INV_SQRT3 * Wvv[(i * 16 + j) * 48 + nt * 16 + kn];
  } else if (region == 2) {                 // W_sv (65,16,16): i = 2c + Kl/16, j = Kl%16
    int i = 2 * cn + (Kl >> 4);
    int j = Kl & 15;
    val = (i < 65) ? Wsv[(i * 16 + j) * 16 + kn] : 0.0f;
  } else {                                  // W_vs (32,32,16): i = c, j = Kl
    val = Wvs[(cn * 32 + Kl) * 16 + kn];
  }
  pk[idx] = (_Float16)val;
}

// ---------------------------------------------------------------------------
// Kernel 1: one wave32 per 16-edge tile; 534 WMMAs per tile.
// ---------------------------------------------------------------------------
__global__ void __launch_bounds__(32) k_main(
    const float* __restrict__ xi, const float* __restrict__ xj,
    const float* __restrict__ ea, const float* __restrict__ el,
    const float* __restrict__ b_s, const _Float16* __restrict__ pk,
    float* __restrict__ scal, float* __restrict__ voutb, float* __restrict__ part)
{
  __shared__ _Float16 ls1[16][68];   // [edge][i] : si | sj | len | zero-pad
  __shared__ _Float16 lv1[16][96];   // [edge][i*3+x], i in [0,32)
  __shared__ _Float16 ls2[16][32];   // [edge][j]
  __shared__ _Float16 lv2[16][48];   // [edge][j*3+x], j in [0,16)
  __shared__ float    lred[80];

  const int lane = threadIdx.x;
  const int tile = blockIdx.x;
  const int e0   = tile * 16;

  // --- stage inputs (f32 -> f16) ---
  for (int t = lane; t < 16 * 80; t += 32) {
    int m = t / 80, c = t - m * 80;
    int g = (e0 + m) * 80 + c;
    _Float16 a = (_Float16)xi[g];
    _Float16 b = (_Float16)xj[g];
    _Float16 w = (_Float16)ea[g];
    if (c < 32) { ls1[m][c] = a; ls1[m][32 + c] = b; ls2[m][c] = w; }
    else        { int r = c - 32; lv1[m][r] = a; lv1[m][48 + r] = b; lv2[m][r] = w; }
  }
  if (lane < 16) {
    ls1[lane][64] = (_Float16)el[e0 + lane];
    ls1[lane][65] = (_Float16)0.0f;
    ls1[lane][66] = (_Float16)0.0f;
    ls1[lane][67] = (_Float16)0.0f;
  }
  for (int t = lane; t < 80; t += 32) lred[t] = 0.0f;
  __syncthreads();

  const int  m  = lane & 15;
  const bool hi = lane >= 16;

  // --- per-lane stationary A-fragment vectors (A layout: h<8 -> K=h(+8 if hi lane),
  //     h>=8 -> K=h+8(+8 if hi lane); j = K%16) ---
  v16h s2f, v2f0, v2f1, v2f2;
#pragma unroll
  for (int h = 0; h < 16; ++h) {
    int Kl = h + ((h < 8) ? 0 : 8) + (hi ? 8 : 0);   // in [0,32)
    s2f[h] = ls2[m][Kl];
    int j = (h & 7) + (hi ? 8 : 0);                  // Kl % 16
    v2f0[h] = lv2[m][j * 3 + 0];
    v2f1[h] = lv2[m][j * 3 + 1];
    v2f2[h] = lv2[m][j * 3 + 2];
  }

  v8f sacc0 = {}, sacc1 = {}, sacc2 = {};
  v8f vacc0 = {}, vacc1 = {}, vacc2 = {};

  const _Float16* ssp = pk + SS_OFF + lane * 16;
  const _Float16* vvp = pk + VV_OFF + lane * 16;
  const _Float16* svp = pk + SV_OFF + lane * 16;
  const _Float16* vsp = pk + VS_OFF + lane * 16;

  // --- term SS: A = s1[i] * s2[j] ---
  for (int c = 0; c < SS_CH; ++c) {
    _Float16 s1c = ls1[m][c];
    v16h a = s2f * s1c;
    v16h b0 = *(const v16h*)(ssp + (c * 3 + 0) * FRAG);
    v16h b1 = *(const v16h*)(ssp + (c * 3 + 1) * FRAG);
    v16h b2 = *(const v16h*)(ssp + (c * 3 + 2) * FRAG);
    sacc0 = WMMA_F16(a, b0, sacc0);
    sacc1 = WMMA_F16(a, b1, sacc1);
    sacc2 = WMMA_F16(a, b2, sacc2);
  }

  // --- term VV: A = v1[i,x] * v2[j,x] (x folded into K; same C accumulators) ---
#pragma unroll
  for (int x = 0; x < 3; ++x) {
    const v16h vf = (x == 0) ? v2f0 : ((x == 1) ? v2f1 : v2f2);
    for (int c16 = 0; c16 < 16; ++c16) {
      int c = x * 16 + c16;
      _Float16 alo = lv1[m][(2 * c16) * 3 + x];
      _Float16 ahi = lv1[m][(2 * c16 + 1) * 3 + x];
      v16h smix;
#pragma unroll
      for (int h = 0; h < 16; ++h) smix[h] = (h < 8) ? alo : ahi;
      v16h a  = smix * vf;
      v16h b0 = *(const v16h*)(vvp + (c * 3 + 0) * FRAG);
      v16h b1 = *(const v16h*)(vvp + (c * 3 + 1) * FRAG);
      v16h b2 = *(const v16h*)(vvp + (c * 3 + 2) * FRAG);
      sacc0 = WMMA_F16(a, b0, sacc0);
      sacc1 = WMMA_F16(a, b1, sacc1);
      sacc2 = WMMA_F16(a, b2, sacc2);
    }
  }

  // --- term SV: A = s1[i] * v2[j,x]; one B reused across the 3 x-accumulators ---
  for (int c = 0; c < SV_CH; ++c) {
    const v16h b = *(const v16h*)(svp + c * FRAG);
    _Float16 slo = ls1[m][2 * c];
    _Float16 shi = ls1[m][2 * c + 1];
    v16h smix;
#pragma unroll
    for (int h = 0; h < 16; ++h) smix[h] = (h < 8) ? slo : shi;
    vacc0 = WMMA_F16(smix * v2f0, b, vacc0);
    vacc1 = WMMA_F16(smix * v2f1, b, vacc1);
    vacc2 = WMMA_F16(smix * v2f2, b, vacc2);
  }

  // --- term VS: A = v1[i,x] * s2[j]; one B reused across the 3 x-accumulators ---
  for (int c = 0; c < VS_CH; ++c) {
    const v16h b = *(const v16h*)(vsp + c * FRAG);
    _Float16 a0 = lv1[m][c * 3 + 0];
    _Float16 a1 = lv1[m][c * 3 + 1];
    _Float16 a2 = lv1[m][c * 3 + 2];
    vacc0 = WMMA_F16(s2f * a0, b, vacc0);
    vacc1 = WMMA_F16(s2f * a1, b, vacc1);
    vacc2 = WMMA_F16(s2f * a2, b, vacc2);
  }

  // --- epilogue: bias, SiLU self-gate, sigmoid vector gate, partial BN sums ---
  // D layout: acc[g] -> edge = e0 + g + (hi?8:0), channel = nt*16 + lane%16.
  const int kb   = lane & 15;
  const int eoff = hi ? 8 : 0;
  const float bs0 = b_s[kb], bs1 = b_s[16 + kb], bs2 = b_s[32 + kb];
  float a_s0 = 0.f, a_s1 = 0.f, a_q0 = 0.f, a_q1 = 0.f, a_v = 0.f;
#pragma unroll
  for (int g = 0; g < 8; ++g) {
    int e = e0 + eoff + g;
    float so0 = sacc0[g] + bs0;
    float so1 = sacc1[g] + bs1;
    float gt  = sacc2[g] + bs2;
    float sg0 = 1.f / (1.f + __expf(-so0));
    float sg1 = 1.f / (1.f + __expf(-so1));
    float gg  = 1.f / (1.f + __expf(-gt));
    float sl0 = so0 * sg0, sl1 = so1 * sg1;
    float v0 = vacc0[g] * gg, v1 = vacc1[g] * gg, v2 = vacc2[g] * gg;
    scal[e * 32 + kb]       = sl0;
    scal[e * 32 + 16 + kb]  = sl1;
    voutb[e * 48 + kb]      = v0;       // layout [e][x][k] for coalescing
    voutb[e * 48 + 16 + kb] = v1;
    voutb[e * 48 + 32 + kb] = v2;
    a_s0 += sl0; a_s1 += sl1;
    a_q0 += sl0 * sl0; a_q1 += sl1 * sl1;
    a_v  += v0 * v0 + v1 * v1 + v2 * v2;
  }
  atomicAdd(&lred[kb],       a_s0);
  atomicAdd(&lred[16 + kb],  a_s1);
  atomicAdd(&lred[32 + kb],  a_q0);
  atomicAdd(&lred[48 + kb],  a_q1);
  atomicAdd(&lred[64 + kb],  a_v);
  __syncthreads();
  for (int t = lane; t < 80; t += 32) part[tile * 80 + t] = lred[t];
}

// ---------------------------------------------------------------------------
// Kernel 2: reduce per-block partials -> BN statistics / scale factors
// stats: [0:32) mu, [32:64) s-scale, [64:80) v-scale
// ---------------------------------------------------------------------------
__global__ void k_stats(const float* __restrict__ part,
                        const float* __restrict__ bn_w_s, const float* __restrict__ bn_w_v,
                        float* __restrict__ stats)
{
  int t = threadIdx.x;
  const float invE = 1.0f / (float)N_EDGES;
  if (t < 32) {
    float s = 0.f, q = 0.f;
    for (int b = 0; b < N_TILES; ++b) {
      s += part[b * 80 + t];
      q += part[b * 80 + 32 + t];
    }
    float mu  = s * invE;
    float var = q * invE - mu * mu;
    stats[t]      = mu;
    stats[32 + t] = bn_w_s[t] * rsqrtf(var + BN_EPS);
  } else if (t < 48) {
    int k = t - 32;
    float v = 0.f;
    for (int b = 0; b < N_TILES; ++b) v += part[b * 80 + 64 + k];
    float vn = v * invE * (1.0f / 3.0f);
    stats[64 + k] = bn_w_v[k] * rsqrtf(vn + BN_EPS);
  }
}

// ---------------------------------------------------------------------------
// Kernel 3: normalize + residual (memory bound)
// ---------------------------------------------------------------------------
__global__ void k_final(const float* __restrict__ scal, const float* __restrict__ voutb,
                        const float* __restrict__ stats, const float* __restrict__ bn_b_s,
                        const float* __restrict__ ea, float* __restrict__ out)
{
  int idx = blockIdx.x * blockDim.x + threadIdx.x;
  if (idx >= N_EDGES * 80) return;
  int e = idx / 80, c = idx - e * 80;
  float r;
  if (c < 32) {
    r = (scal[e * 32 + c] - stats[c]) * stats[32 + c] + bn_b_s[c];
  } else {
    int rr = c - 32;
    int k = rr / 3, x = rr - 3 * k;
    r = voutb[e * 48 + x * 16 + k] * stats[64 + k];
  }
  out[idx] = r + ea[idx];
}

// ---------------------------------------------------------------------------
extern "C" void kernel_launch(void* const* d_in, const int* in_sizes, int n_in,
                              void* d_out, int out_size, void* d_ws, size_t ws_size,
                              hipStream_t stream)
{
  (void)in_sizes; (void)n_in; (void)out_size; (void)ws_size;
  const float* xi     = (const float*)d_in[0];
  const float* xj     = (const float*)d_in[1];
  const float* ea     = (const float*)d_in[2];
  const float* el     = (const float*)d_in[3];
  const float* Wss    = (const float*)d_in[4];
  const float* Wvv    = (const float*)d_in[5];
  const float* Wsv    = (const float*)d_in[6];
  const float* Wvs    = (const float*)d_in[7];
  const float* b_s    = (const float*)d_in[8];
  const float* bn_w_s = (const float*)d_in[9];
  const float* bn_w_v = (const float*)d_in[10];
  const float* bn_b_s = (const float*)d_in[11];

  char* ws = (char*)d_ws;
  _Float16* pk   = (_Float16*)ws;
  float* scal    = (float*)(ws + OFF_SCAL);
  float* voutb   = (float*)(ws + OFF_VOUT);
  float* part    = (float*)(ws + OFF_PART);
  float* stats   = (float*)(ws + OFF_STATS);

  k_pack <<<(PK_TOTAL + 255) / 256, 256, 0, stream>>>(Wss, Wvv, Wsv, Wvs, pk);
  k_main <<<N_TILES, 32, 0, stream>>>(xi, xj, ea, el, b_s, pk, scal, voutb, part);
  k_stats<<<1, 64, 0, stream>>>(part, bn_w_s, bn_w_v, stats);
  k_final<<<(N_EDGES * 80 + 255) / 256, 256, 0, stream>>>(scal, voutb, stats, bn_b_s, ea,
                                                          (float*)d_out);
}
